// IntronsDecoder_54743653154969
// MI455X (gfx1250) — matrix operand order repaired and checked by reading, not compiled, not run on hardware
//
#include <hip/hip_runtime.h>
#include <hip/hip_bf16.h>
#include <cstdint>

#define B_SZ   4096
#define N_IN   32
#define N_HID  128
#define N_OUT  30000
#define BN_EPS 1e-3f

#define TM 128          // row tile
#define TN 48           // col tile = 3 * 16 -> clusters are tile-local
#define LDS_STRIDE 50   // 48 + 2 pad dwords: conflict-free for D-layout stores

typedef __attribute__((ext_vector_type(16))) __bf16 v16bf;
typedef __attribute__((ext_vector_type(8)))  float  v8f;

static __device__ __forceinline__ uint16_t f32_to_bf16(float f) {
  union { float f; uint32_t u; } c; c.f = f;
  uint32_t u = c.u;
  u += 0x7FFFu + ((u >> 16) & 1u);   // round-to-nearest-even
  return (uint16_t)(u >> 16);
}

// ---------------- K1: h = z @ W1^T + b1 ----------------
__global__ void k1_layer1(const float* __restrict__ z, const float* __restrict__ W1,
                          const float* __restrict__ b1, float* __restrict__ h) {
  __shared__ float zrow[N_IN];
  const int b = blockIdx.x;
  const int j = threadIdx.x;                 // 128 threads
  if (j < N_IN) zrow[j] = z[b * N_IN + j];
  __syncthreads();
  float acc = b1[j];
  const float* w = W1 + j * N_IN;
  #pragma unroll
  for (int k = 0; k < N_IN; ++k) acc = fmaf(zrow[k], w[k], acc);
  h[b * N_HID + j] = acc;
}

// ---------------- K2: batch mean/var -> scale/shift (deterministic tree) ----
__global__ void k2_stats(const float* __restrict__ h, const float* __restrict__ gamma,
                         const float* __restrict__ beta, float* __restrict__ scale,
                         float* __restrict__ shift) {
  __shared__ float ssum[256];
  __shared__ float ssq[256];
  const int j = blockIdx.x;                  // hidden unit
  const int t = threadIdx.x;
  float s = 0.f, q = 0.f;
  for (int b = t; b < B_SZ; b += 256) {
    float v = h[b * N_HID + j];
    s += v; q += v * v;
  }
  ssum[t] = s; ssq[t] = q;
  __syncthreads();
  for (int off = 128; off > 0; off >>= 1) {
    if (t < off) { ssum[t] += ssum[t + off]; ssq[t] += ssq[t + off]; }
    __syncthreads();
  }
  if (t == 0) {
    float mu  = ssum[0] * (1.0f / (float)B_SZ);
    float var = ssq[0] * (1.0f / (float)B_SZ) - mu * mu;
    float sc  = gamma[j] * rsqrtf(var + BN_EPS);
    scale[j] = sc;
    shift[j] = beta[j] - mu * sc;
  }
}

// ---------------- K3: BN + ReLU + cvt bf16 ----------------
__global__ void k3_bn_relu_cvt(const float* __restrict__ h, const float* __restrict__ scale,
                               const float* __restrict__ shift, uint16_t* __restrict__ hb) {
  const int idx = blockIdx.x * blockDim.x + threadIdx.x;
  const int j = idx & (N_HID - 1);
  float v = fmaf(h[idx], scale[j], shift[j]);
  hb[idx] = f32_to_bf16(fmaxf(v, 0.0f));
}

// ---------------- K4: W2 -> bf16 ----------------
__global__ void k4_cvt_w2(const float* __restrict__ W2, uint16_t* __restrict__ w2b, int n) {
  const int idx = blockIdx.x * blockDim.x + threadIdx.x;
  if (idx < n) w2b[idx] = f32_to_bf16(W2[idx]);
}

// ---------------- K5: WMMA GEMM + fused cluster softmax ----------------
__global__ __launch_bounds__(256)
void k5_gemm_softmax(const uint16_t* __restrict__ hb, const uint16_t* __restrict__ w2b,
                     const float* __restrict__ b2, float* __restrict__ out) {
  __shared__ float spot[TM * LDS_STRIDE];

  const int tid  = threadIdx.x;
  const int wave = tid >> 5;                 // wave32: 8 waves per block
  const int lane = tid & 31;
  const int g    = lane >> 4;                // half-wave select
  const int n16  = lane & 15;

  const int rowBase = blockIdx.y * TM;       // 32 row blocks
  const int colBase = blockIdx.x * TN;       // 625 col blocks

  // A fragment rows: ISA 16-bit A 16x32 layout -> M = lane%16, K split by g
  const uint16_t* aPtr = hb + (size_t)(rowBase + wave * 16 + n16) * N_HID;
  // B fragment: lane holds column n = colBase + t*16 + n16, 16 sequential K per lane
  const uint16_t* bPtr = w2b + (size_t)(colBase + n16) * N_HID;

  v8f acc0 = {}, acc1 = {}, acc2 = {};

  #pragma unroll
  for (int kk = 0; kk < 4; ++kk) {
    const int koff = kk * 32;
    union { v16bf v; uint4 u[2]; } af, bf0, bf1, bf2;
    // A: halves 0..15 = K[koff+16g .. +15] for this lane's row
    af.u[0] = *(const uint4*)(aPtr + koff + 16 * g);
    af.u[1] = *(const uint4*)(aPtr + koff + 16 * g + 8);
    // B: halves 0..15 = K[koff+16g .. +15] for this lane's column
    const uint16_t* bk = bPtr + koff + 16 * g;
    bf0.u[0] = *(const uint4*)(bk);
    bf0.u[1] = *(const uint4*)(bk + 8);
    bf1.u[0] = *(const uint4*)(bk + 16 * N_HID);
    bf1.u[1] = *(const uint4*)(bk + 16 * N_HID + 8);
    bf2.u[0] = *(const uint4*)(bk + 32 * N_HID);
    bf2.u[1] = *(const uint4*)(bk + 32 * N_HID + 8);

    acc0 = __builtin_amdgcn_wmma_f32_16x16x32_bf16(false, af.v, false, bf0.v,
                                                   (short)0, acc0, false, false);
    acc1 = __builtin_amdgcn_wmma_f32_16x16x32_bf16(false, af.v, false, bf1.v,
                                                   (short)0, acc1, false, false);
    acc2 = __builtin_amdgcn_wmma_f32_16x16x32_bf16(false, af.v, false, bf2.v,
                                                   (short)0, acc2, false, false);
  }

  // D layout: VGPR r, lane -> M = r + 8*g, N = lane%16
  {
    const int mBase = wave * 16 + 8 * g;
    #pragma unroll
    for (int r = 0; r < 8; ++r) {
      float* row = &spot[(mBase + r) * LDS_STRIDE];
      row[n16]      = acc0[r];
      row[16 + n16] = acc1[r];
      row[32 + n16] = acc2[r];
    }
  }
  __syncthreads();

  // Epilogue: 128 rows * 16 clusters = 2048 tasks; cluster col0 has pot==0 -> e0==1
  #pragma unroll
  for (int it = 0; it < 8; ++it) {
    const int task = tid + it * 256;
    const int rr = task >> 4;                // local row 0..127
    const int c  = task & 15;                // local cluster 0..15
    const float* p = &spot[rr * LDS_STRIDE + c * 3];
    const int col = colBase + c * 3;         // global col, col%3==0
    float e1 = expf(p[1] + b2[col + 1]);
    float e2 = expf(p[2] + b2[col + 2]);
    float inv = 1.0f / (1.0f + e1 + e2);
    const size_t o = (size_t)(rowBase + rr) * N_OUT + col;
    out[o]     = inv;                        // exp(0)/sum
    out[o + 1] = e1 * inv;
    out[o + 2] = e2 * inv;
  }
}

extern "C" void kernel_launch(void* const* d_in, const int* in_sizes, int n_in,
                              void* d_out, int out_size, void* d_ws, size_t ws_size,
                              hipStream_t stream) {
  const float* z     = (const float*)d_in[0];
  const float* W1    = (const float*)d_in[1];
  const float* b1    = (const float*)d_in[2];
  const float* gamma = (const float*)d_in[3];
  const float* beta  = (const float*)d_in[4];
  const float* W2    = (const float*)d_in[5];
  const float* b2    = (const float*)d_in[6];
  // d_in[7] intron_clusters (= j/3) and d_in[8] first_indices (= 3c) are implicit.
  float* out = (float*)d_out;

  char* ws = (char*)d_ws;
  size_t off = 0;
  float*    h     = (float*)(ws + off);    off += (size_t)B_SZ * N_HID * 4;   // 2 MB
  float*    scale = (float*)(ws + off);    off += 256 * 4;
  float*    shift = (float*)(ws + off);    off += 256 * 4;
  uint16_t* hbb   = (uint16_t*)(ws + off); off += (size_t)B_SZ * N_HID * 2;   // 1 MB
  uint16_t* w2b   = (uint16_t*)(ws + off); off += (size_t)N_OUT * N_HID * 2;  // 7.68 MB

  k1_layer1<<<B_SZ, N_HID, 0, stream>>>(z, W1, b1, h);
  k2_stats<<<N_HID, 256, 0, stream>>>(h, gamma, beta, scale, shift);
  k3_bn_relu_cvt<<<(B_SZ * N_HID) / 256, 256, 0, stream>>>(h, scale, shift, hbb);
  k4_cvt_w2<<<(N_OUT * N_HID + 255) / 256, 256, 0, stream>>>(W2, w2b, N_OUT * N_HID);

  dim3 grid(N_OUT / TN, B_SZ / TM);          // 625 x 32 blocks, 256 threads (8 waves)
  k5_gemm_softmax<<<grid, 256, 0, stream>>>(hbb, w2b, b2, out);
}
